// SelfAttention_79276506350285
// MI455X (gfx1250) — matrix operand order repaired and checked by reading, not compiled, not run on hardware
//
#include <hip/hip_runtime.h>

#define HEADS 16
#define DIMH 64
#define SEQ 2048
#define DMODEL 1024
#define BATCH 2

typedef __attribute__((ext_vector_type(16))) __bf16 v16bf;
typedef __attribute__((ext_vector_type(2)))  __bf16 v2bf;
typedef __attribute__((ext_vector_type(8)))  float  v8f;
typedef __attribute__((ext_vector_type(4)))  unsigned u32x4;

__device__ __forceinline__ __bf16 f2bf(float f) { return (__bf16)f; }

union FragU { v16bf v; u32x4 q[2]; };
union Pack8 { u32x4 q; v2bf p[4]; };

__device__ __forceinline__ v16bf ld_frag(const __bf16* p0, const __bf16* p1) {
  FragU u;
  u.q[0] = *(const u32x4*)p0;
  u.q[1] = *(const u32x4*)p1;
  return u.v;
}

#define WMMA_BF16(A, B, C) \
  __builtin_amdgcn_wmma_f32_16x16x32_bf16(false, (A), false, (B), (short)0, (C), false, false)

// ---------------------------------------------------------------------------
// Prep 1: x (fp32) -> xb (bf16)
// ---------------------------------------------------------------------------
__global__ void __launch_bounds__(256)
x_cvt_kernel(const float* __restrict__ x, __bf16* __restrict__ xb) {
  size_t i = ((size_t)blockIdx.x * blockDim.x + threadIdx.x) * 2;
  v2bf p;
  p[0] = f2bf(x[i]);
  p[1] = f2bf(x[i + 1]);
  *(v2bf*)(xb + i) = p;
}

// ---------------------------------------------------------------------------
// Prep 2: W (fp32, [K][C]) -> Wt (bf16, [C][K]) via LDS tile transpose
// ---------------------------------------------------------------------------
__global__ void __launch_bounds__(256)
w_cvt_transpose_kernel(const float* __restrict__ W, __bf16* __restrict__ Wt,
                       int K, int C) {
  __shared__ __bf16 t[32][33];
  const int c0 = blockIdx.x * 32;
  const int k0 = blockIdx.y * 32;
  const int tx = threadIdx.x & 31;
  const int ty = threadIdx.x >> 5;
#pragma unroll
  for (int kk = ty; kk < 32; kk += 8)
    t[kk][tx] = f2bf(W[(size_t)(k0 + kk) * C + c0 + tx]);
  __syncthreads();
#pragma unroll
  for (int cc = ty; cc < 32; cc += 8)
    Wt[(size_t)(c0 + cc) * K + k0 + tx] = t[tx][cc];
}

// ---------------------------------------------------------------------------
// Kernel 1: qk = xb @ Wqkt^T, 32x32 per wave (4 WMMA / 8 loads per k-step).
// Writes q (beta-folded, [b,h,n,d]), k ([b,h,n,d]), kT ([b,h,d,n]).
// ---------------------------------------------------------------------------
__global__ void __launch_bounds__(128)
qk_proj_kernel(const __bf16* __restrict__ xb, const __bf16* __restrict__ Wqkt,
               __bf16* __restrict__ qb, __bf16* __restrict__ kbuf,
               __bf16* __restrict__ kT) {
  const int lane = threadIdx.x & 31;
  const int wv   = threadIdx.x >> 5;
  const int mt   = blockIdx.y;              // 0..127
  const int nt   = blockIdx.x * 4 + wv;     // 0..63
  const int R0   = mt * 32;
  const int C0   = nt * 32;
  const int m    = lane & 15;
  const int hi   = lane >> 4;
  const int kbA  = hi << 4;

  const __bf16* arow0 = xb + (size_t)(R0 + m) * DMODEL;
  const __bf16* arow1 = arow0 + (size_t)16 * DMODEL;
  const __bf16* bcol0 = Wqkt + (size_t)(C0 + m) * DMODEL + 16 * hi;
  const __bf16* bcol1 = bcol0 + (size_t)16 * DMODEL;

  v8f acc00 = {}, acc01 = {}, acc10 = {}, acc11 = {};
  for (int kk = 0; kk < DMODEL; kk += 32) {
    v16bf a0 = ld_frag(arow0 + kk + kbA, arow0 + kk + 8 + kbA);
    v16bf a1 = ld_frag(arow1 + kk + kbA, arow1 + kk + 8 + kbA);
    v16bf b0 = ld_frag(bcol0 + kk, bcol0 + kk + 8);
    v16bf b1 = ld_frag(bcol1 + kk, bcol1 + kk + 8);
    acc00 = WMMA_BF16(a0, b0, acc00);
    acc01 = WMMA_BF16(a0, b1, acc01);
    acc10 = WMMA_BF16(a1, b0, acc10);
    acc11 = WMMA_BF16(a1, b1, acc11);
  }

  const float beta = 0.125f;  // 1/sqrt(64)
  const int bidx = R0 >> 11;
  const int nb   = (R0 & (SEQ - 1)) + 8 * hi;
  v8f accs[2][2] = {{acc00, acc01}, {acc10, acc11}};

  if (C0 < DMODEL) {  // whole 32-wide tile is q
#pragma unroll
    for (int ch = 0; ch < 2; ++ch) {
      int col = C0 + 16 * ch + m;
      int h = col >> 6, d = col & 63;
#pragma unroll
      for (int rh = 0; rh < 2; ++rh) {
        __bf16* qp = qb + (((size_t)(bidx * HEADS + h)) * SEQ + nb + 16 * rh) * DIMH + d;
        v8f a = accs[rh][ch];
#pragma unroll
        for (int r = 0; r < 8; ++r) qp[(size_t)r * DIMH] = f2bf(a[r] * beta);
      }
    }
  } else {            // whole tile is k: write k and kT
#pragma unroll
    for (int ch = 0; ch < 2; ++ch) {
      int c2 = C0 + 16 * ch + m - DMODEL;
      int h = c2 >> 6, d = c2 & 63;
#pragma unroll
      for (int rh = 0; rh < 2; ++rh) {
        v8f a = accs[rh][ch];
        __bf16* kp = kbuf + (((size_t)(bidx * HEADS + h)) * SEQ + nb + 16 * rh) * DIMH + d;
#pragma unroll
        for (int r = 0; r < 8; ++r) kp[(size_t)r * DIMH] = f2bf(a[r]);
        Pack8 u;
#pragma unroll
        for (int r2 = 0; r2 < 4; ++r2) {
          u.p[r2][0] = f2bf(a[2 * r2]);
          u.p[r2][1] = f2bf(a[2 * r2 + 1]);
        }
        *(u32x4*)(kT + (((size_t)(bidx * HEADS + h)) * DIMH + d) * SEQ + nb + 16 * rh) = u.q;
      }
    }
  }
}

// ---------------------------------------------------------------------------
// Kernel 2: causal flash attention, K reused as V.
// One wave = 32 query rows; 32-wide K/V tiles.
// Row-sums (softmax denominator) computed on the matrix pipe as P @ 1,
// sharing the alpha-rescaled accumulator path with the output.
// ---------------------------------------------------------------------------
__global__ void __launch_bounds__(128)
attn_kernel(const __bf16* __restrict__ qb, const __bf16* __restrict__ kb,
            const __bf16* __restrict__ kT, __bf16* __restrict__ attn) {
  __shared__ __bf16 Plds[4][32 * 32];

  const int lane = threadIdx.x & 31;
  const int wv   = threadIdx.x >> 5;
  const int bh   = blockIdx.y;               // 0..31
  const int it   = blockIdx.x * 4 + wv;      // 0..63
  const int R0   = it * 32;
  const int m    = lane & 15;
  const int hi   = lane >> 4;
  const int kbA  = hi << 4;

  const __bf16* q  = qb + (size_t)bh * SEQ * DIMH;
  const __bf16* k  = kb + (size_t)bh * SEQ * DIMH;
  const __bf16* kt = kT + (size_t)bh * DIMH * SEQ;

  const __bf16* qr0 = q + (size_t)(R0 + m) * DIMH;
  const __bf16* qr1 = qr0 + (size_t)16 * DIMH;
  v16bf aq[2][2];
  aq[0][0] = ld_frag(qr0 + kbA,      qr0 + 8 + kbA);
  aq[0][1] = ld_frag(qr0 + 32 + kbA, qr0 + 40 + kbA);
  aq[1][0] = ld_frag(qr1 + kbA,      qr1 + 8 + kbA);
  aq[1][1] = ld_frag(qr1 + 32 + kbA, qr1 + 40 + kbA);

  // all-ones B fragment: row-sum generator for P @ 1
  v16bf ones;
#pragma unroll
  for (int e = 0; e < 16; ++e) ones[e] = f2bf(1.0f);

  v8f acc[2][4];       // output accumulators
  v8f accl[2];         // row-sum (softmax denominator) accumulators
  float mrow[2][8];
#pragma unroll
  for (int rh = 0; rh < 2; ++rh) {
    accl[rh] = (v8f){};
#pragma unroll
    for (int r = 0; r < 8; ++r) mrow[rh][r] = -3.0e38f;
#pragma unroll
    for (int nf = 0; nf < 4; ++nf) acc[rh][nf] = (v8f){};
  }

  const int jmax = R0 >> 5;   // diagonal 32-col tile
  for (int jt = 0; jt <= jmax; ++jt) {
    const int C0 = jt * 32;

    // ---- S = q @ k^T : 4 shared B fragments feed 8 WMMAs ----
    const __bf16* r0p = k + (size_t)(C0 + m) * DIMH + 16 * hi;
    const __bf16* r1p = r0p + 16 * DIMH;
    v16bf bk0a = ld_frag(r0p,      r0p + 8);
    v16bf bk0b = ld_frag(r0p + 32, r0p + 40);
    v16bf bk1a = ld_frag(r1p,      r1p + 8);
    v16bf bk1b = ld_frag(r1p + 32, r1p + 40);
    v8f s[2][2];
#pragma unroll
    for (int rh = 0; rh < 2; ++rh) {
      v8f t0 = {}, t1 = {};
      t0 = WMMA_BF16(aq[rh][0], bk0a, t0);
      t0 = WMMA_BF16(aq[rh][1], bk0b, t0);
      t1 = WMMA_BF16(aq[rh][0], bk1a, t1);
      t1 = WMMA_BF16(aq[rh][1], bk1b, t1);
      s[rh][0] = t0;
      s[rh][1] = t1;
    }

    // ---- causal mask: only the diagonal tile is partial ----
    if (jt == jmax) {
#pragma unroll
      for (int rh = 0; rh < 2; ++rh)
#pragma unroll
        for (int r = 0; r < 8; ++r) {
          int row = R0 + 16 * rh + r + 8 * hi;
          int c0  = C0 + m;
          if (c0 > row)      s[rh][0][r] = -3.0e38f;
          if (c0 + 16 > row) s[rh][1][r] = -3.0e38f;
        }
    }

    // ---- online softmax (max via shuffles; sum deferred to P @ 1) ----
    __bf16* pl = Plds[wv];
#pragma unroll
    for (int rh = 0; rh < 2; ++rh) {
#pragma unroll
      for (int r = 0; r < 8; ++r) {
        float mx = fmaxf(s[rh][0][r], s[rh][1][r]);
#pragma unroll
        for (int d = 1; d < 16; d <<= 1) mx = fmaxf(mx, __shfl_xor(mx, d, 32));
        float mnew  = fmaxf(mrow[rh][r], mx);
        float alpha = __expf(mrow[rh][r] - mnew);
        mrow[rh][r] = mnew;
        float p0 = __expf(s[rh][0][r] - mnew);
        float p1 = __expf(s[rh][1][r] - mnew);
        acc[rh][0][r] *= alpha; acc[rh][1][r] *= alpha;
        acc[rh][2][r] *= alpha; acc[rh][3][r] *= alpha;
        accl[rh][r]   *= alpha;
        int rl = 16 * rh + r + 8 * hi;
        pl[rl * 32 + m]      = f2bf(p0);
        pl[rl * 32 + 16 + m] = f2bf(p1);
      }
    }
    asm volatile("s_wait_dscnt 0" ::: "memory");

    // ---- P fragments for both row-halves ----
    v16bf ap0 = ld_frag(pl + m * 32 + kbA,        pl + m * 32 + 8 + kbA);
    v16bf ap1 = ld_frag(pl + (16 + m) * 32 + kbA, pl + (16 + m) * 32 + 8 + kbA);

    // ---- row sums on the matrix pipe: l += P @ 1 ----
    accl[0] = WMMA_BF16(ap0, ones, accl[0]);
    accl[1] = WMMA_BF16(ap1, ones, accl[1]);

    // ---- out += P @ V via kT: 4 shared V fragments feed 8 WMMAs ----
    const __bf16* vbase = kt + (size_t)m * SEQ + C0 + 16 * hi;
#pragma unroll
    for (int nf = 0; nf < 4; ++nf) {
      const __bf16* vp = vbase + (size_t)(16 * nf) * SEQ;
      v16bf bv = ld_frag(vp, vp + 8);
      acc[0][nf] = WMMA_BF16(ap0, bv, acc[0][nf]);
      acc[1][nf] = WMMA_BF16(ap1, bv, acc[1][nf]);
    }
  }

  // ---- normalize, store bf16 in [b, n, h*d] ----
  const int bidx = bh >> 4;
  const int h    = bh & 15;
#pragma unroll
  for (int rh = 0; rh < 2; ++rh)
#pragma unroll
    for (int r = 0; r < 8; ++r) {
      int n = R0 + 16 * rh + r + 8 * hi;
      float inv = 1.0f / accl[rh][r];
      __bf16* op = attn + ((size_t)(bidx * SEQ + n)) * DMODEL + h * DIMH + m;
      op[0]  = f2bf(acc[rh][0][r] * inv);
      op[16] = f2bf(acc[rh][1][r] * inv);
      op[32] = f2bf(acc[rh][2][r] * inv);
      op[48] = f2bf(acc[rh][3][r] * inv);
    }
}

// ---------------------------------------------------------------------------
// Kernel 3: out = attn(bf16) @ Woutt^T -> fp32, 32x32 per wave
// ---------------------------------------------------------------------------
__global__ void __launch_bounds__(128)
out_proj_kernel(const __bf16* __restrict__ attn, const __bf16* __restrict__ Woutt,
                float* __restrict__ out) {
  const int lane = threadIdx.x & 31;
  const int wv   = threadIdx.x >> 5;
  const int mt   = blockIdx.y;              // 0..127
  const int nt   = blockIdx.x * 4 + wv;     // 0..31
  const int R0   = mt * 32;
  const int C0   = nt * 32;
  const int m    = lane & 15;
  const int hi   = lane >> 4;
  const int kbA  = hi << 4;

  const __bf16* arow0 = attn + (size_t)(R0 + m) * DMODEL;
  const __bf16* arow1 = arow0 + (size_t)16 * DMODEL;
  const __bf16* bcol0 = Woutt + (size_t)(C0 + m) * DMODEL + 16 * hi;
  const __bf16* bcol1 = bcol0 + (size_t)16 * DMODEL;

  v8f acc00 = {}, acc01 = {}, acc10 = {}, acc11 = {};
  for (int kk = 0; kk < DMODEL; kk += 32) {
    v16bf a0 = ld_frag(arow0 + kk + kbA, arow0 + kk + 8 + kbA);
    v16bf a1 = ld_frag(arow1 + kk + kbA, arow1 + kk + 8 + kbA);
    v16bf b0 = ld_frag(bcol0 + kk, bcol0 + kk + 8);
    v16bf b1 = ld_frag(bcol1 + kk, bcol1 + kk + 8);
    acc00 = WMMA_BF16(a0, b0, acc00);
    acc01 = WMMA_BF16(a0, b1, acc01);
    acc10 = WMMA_BF16(a1, b0, acc10);
    acc11 = WMMA_BF16(a1, b1, acc11);
  }

  v8f accs[2][2] = {{acc00, acc01}, {acc10, acc11}};
#pragma unroll
  for (int rh = 0; rh < 2; ++rh)
#pragma unroll
    for (int ch = 0; ch < 2; ++ch) {
      v8f a = accs[rh][ch];
#pragma unroll
      for (int r = 0; r < 8; ++r)
        out[(size_t)(R0 + 16 * rh + r + 8 * hi) * DMODEL + C0 + 16 * ch + m] = a[r];
    }
}

// ---------------------------------------------------------------------------
extern "C" void kernel_launch(void* const* d_in, const int* in_sizes, int n_in,
                              void* d_out, int out_size, void* d_ws, size_t ws_size,
                              hipStream_t stream) {
  (void)in_sizes; (void)n_in; (void)out_size; (void)ws_size;
  const float* x    = (const float*)d_in[0];
  const float* Wqk  = (const float*)d_in[1];
  const float* Wout = (const float*)d_in[2];
  float* out = (float*)d_out;

  const size_t tok   = (size_t)BATCH * SEQ;                 // 4096
  const size_t qk_el = (size_t)BATCH * HEADS * SEQ * DIMH;  // 4M

  __bf16* xb    = (__bf16*)d_ws;                       //  8 MB
  __bf16* Wqkt  = xb    + tok * DMODEL;                //  4 MB
  __bf16* Woutt = Wqkt  + (size_t)2 * DMODEL * DMODEL; //  2 MB
  __bf16* qb    = Woutt + (size_t)DMODEL * DMODEL;     //  8 MB
  __bf16* kb    = qb    + qk_el;                       //  8 MB
  __bf16* kT    = kb    + qk_el;                       //  8 MB
  __bf16* attn  = kT    + qk_el;                       //  8 MB (46 MB total)

  x_cvt_kernel<<<dim3((unsigned)(tok * DMODEL / 2 / 256)), dim3(256), 0, stream>>>(x, xb);
  w_cvt_transpose_kernel<<<dim3(2 * DMODEL / 32, DMODEL / 32), dim3(256), 0, stream>>>(
      Wqk, Wqkt, DMODEL, 2 * DMODEL);
  w_cvt_transpose_kernel<<<dim3(DMODEL / 32, DMODEL / 32), dim3(256), 0, stream>>>(
      Wout, Woutt, DMODEL, DMODEL);

  dim3 blk(128, 1, 1);
  qk_proj_kernel<<<dim3(16, 128, 1), blk, 0, stream>>>(xb, Wqkt, qb, kb, kT);
  attn_kernel  <<<dim3(16, 32, 1),  blk, 0, stream>>>(qb, kb, kT, attn);
  out_proj_kernel<<<dim3(8, 128, 1), blk, 0, stream>>>(attn, Woutt, out);
}
